// LM_7060926235169
// MI455X (gfx1250) — matrix operand order repaired and checked by reading, not compiled, not run on hardware
//
#include <hip/hip_runtime.h>
#include <hip/hip_bf16.h>
#include <cstdint>
#include <cstddef>

// ---------------------------------------------------------------------------
// Model constants (match the reference)
// ---------------------------------------------------------------------------
#define SEQ   2048
#define DM    512
#define NL    8
#define NH    8
#define VOC   50257
#define VOCP  50304          // 1572*32: padded row count for transposed LM weight
#define EPSLN 1e-5f

typedef __attribute__((ext_vector_type(16))) __bf16 v16bf;
typedef __attribute__((ext_vector_type(8)))  __bf16 v8bf;
typedef __attribute__((ext_vector_type(8)))  float  v8f;

__device__ __forceinline__ __bf16 f2bf(float x) {
    union { float f; unsigned int u; } in; in.f = x;
    unsigned int u = in.u;
    unsigned int r = (u + 0x7FFFu + ((u >> 16) & 1u)) >> 16;  // round-to-nearest-even
    union { unsigned short s; __bf16 b; } out; out.s = (unsigned short)r;
    return out.b;
}

// One 16B async global->LDS copy (CDNA5 GLOBAL_LOAD_ASYNC_TO_LDS_B128, GV
// addressing, tracked by ASYNCcnt). ldsOff = LDS byte address (low 32 bits of
// the generic address of a __shared__ object, per the aperture truncation rule).
__device__ __forceinline__ void async_copy16(unsigned ldsOff, const void* gaddr) {
    unsigned long long ga = (unsigned long long)(uintptr_t)gaddr;
    asm volatile("global_load_async_to_lds_b128 %0, %1, off"
                 :: "v"(ldsOff), "v"(ga) : "memory");
}
__device__ __forceinline__ void async_wait0() {
    asm volatile("s_wait_asynccnt 0x0" ::: "memory");
}

// ---------------------------------------------------------------------------
// Tiled bf16 WMMA GEMM:  out = alpha * A(MxK) @ B(KxN) [+ bias][+ resid][swish]
//   A  row-major bf16 [M,K] (lda)
//   Bt row-major bf16 [N,K] (ldb)  -> B[k,n] = Bt[n*ldb + k]  (B pre-transposed)
//   outF (f32, ldc) and/or outB (bf16, ldcb); resid is f32 with pitch ldc.
//   causal!=0 -> skip blocks entirely above the diagonal (never read downstream).
//
// Block tile 128x128, K-step 32, 8 waves; wave (wm in 0..1, wn in 0..3) owns a
// 64x32 tile = 4x2 fragments of v_wmma_f32_16x16x32_bf16.
// LDS: double-buffered sA[m][k], sB[n][k], pitch 40 halves ->
//      every fragment = 2x ds_load_b128; stage k+1 streams in via
//      global_load_async_to_lds_b128 while WMMAs consume stage k.
// Rows/cols beyond M/N are not staged; the stale LDS they'd feed only reaches
// output elements the guarded epilogue never stores.
// ---------------------------------------------------------------------------
__global__ __launch_bounds__(256, 2)
void gemm_bf16(const __bf16* __restrict__ A, const __bf16* __restrict__ Bt,
               float* __restrict__ outF, __bf16* __restrict__ outB,
               const float* __restrict__ bias, const float* __restrict__ resid,
               int M, int N, int K, int lda, int ldb, int ldc, int ldcb,
               float alpha, int swish, int causal)
{
    constexpr int PITCH = 40;        // 32 + 8 halves: 16B-aligned rows, skewed banks
    constexpr int BUF   = 128 * PITCH;
    __shared__ __align__(16) __bf16 sA[2 * BUF];
    __shared__ __align__(16) __bf16 sB[2 * BUF];

    const int bm0 = blockIdx.y * 128;
    const int bn0 = blockIdx.x * 128;
    if (causal && bn0 > bm0 + 127) return;

    const int tid  = threadIdx.x;
    const int lane = tid & 31;
    const int wid  = tid >> 5;
    const int wm   = wid >> 2;        // 0..1
    const int wn   = wid & 3;         // 0..3
    const int lrow = lane & 15;
    const int lhi  = lane >> 4;       // 0/1
    const int kbA  = lhi * 8;         // A frag: K runs [kbA, kbA+7] and [kbA+16, kbA+23]
    const int kbB  = lhi * 8;         // B frag: K runs [kbB, kbB+7] and [kbB+16, kbB+23]

    // 128 rows x 32 k tile -> 2 async b128 per thread per matrix
    auto stage = [&](const __bf16* __restrict__ src, int ld, int base, int rows,
                     __bf16* dst, int k0, int buf) {
        #pragma unroll
        for (int it = 0; it < 2; ++it) {
            int c   = tid + it * 256;          // 0..511
            int row = c >> 2;                  // 0..127
            int kc  = (c & 3) * 8;             // 0,8,16,24
            int g   = base + row;
            if (g < rows)
                async_copy16((unsigned)(uintptr_t)&dst[buf * BUF + row * PITCH + kc],
                             src + (size_t)g * ld + (k0 + kc));
        }
    };

    v8f acc[4][2];
    #pragma unroll
    for (int i = 0; i < 4; ++i)
        #pragma unroll
        for (int j = 0; j < 2; ++j)
            #pragma unroll
            for (int r = 0; r < 8; ++r)
                acc[i][j][r] = 0.0f;

    // ---- prologue: stage 0 into buffer 0 ----
    stage(A,  lda, bm0, M, sA, 0, 0);
    stage(Bt, ldb, bn0, N, sB, 0, 0);
    async_wait0();
    __syncthreads();

    for (int k0 = 0; k0 < K; k0 += 32) {
        const int  cur  = (k0 >> 5) & 1;
        const int  nxt  = cur ^ 1;
        const bool more = (k0 + 32) < K;

        // stream stage k+1 while WMMAs consume stage k
        if (more) {
            stage(A,  lda, bm0, M, sA, k0 + 32, nxt);
            stage(Bt, ldb, bn0, N, sB, k0 + 32, nxt);
        }

        // ---- fragments + 8 WMMAs on buffer `cur` ----
        v16bf af[4], bfr[2];
        #pragma unroll
        for (int i = 0; i < 4; ++i) {
            int m = wm * 64 + i * 16 + lrow;
            union { v16bf v; v8bf h[2]; } u;
            u.h[0] = *(const v8bf*)(&sA[cur * BUF + m * PITCH + kbA]);
            u.h[1] = *(const v8bf*)(&sA[cur * BUF + m * PITCH + kbA + 16]);
            af[i] = u.v;
        }
        #pragma unroll
        for (int j = 0; j < 2; ++j) {
            int n = wn * 32 + j * 16 + lrow;
            union { v16bf v; v8bf h[2]; } u;
            u.h[0] = *(const v8bf*)(&sB[cur * BUF + n * PITCH + kbB]);
            u.h[1] = *(const v8bf*)(&sB[cur * BUF + n * PITCH + kbB + 16]);
            bfr[j] = u.v;
        }
        #pragma unroll
        for (int i = 0; i < 4; ++i)
            #pragma unroll
            for (int j = 0; j < 2; ++j)
                acc[i][j] = __builtin_amdgcn_wmma_f32_16x16x32_bf16(
                    false, af[i], false, bfr[j], (short)0, acc[i][j], false, false);

        if (more) async_wait0();   // our stage-(k+1) copies have landed
        __syncthreads();           // everyone's have
    }

    // ---- epilogue: D-tile element (r, lane) -> m = lhi*8 + r, n = lrow ----
    #pragma unroll
    for (int i = 0; i < 4; ++i) {
        const int mbase = bm0 + wm * 64 + i * 16 + lhi * 8;
        #pragma unroll
        for (int j = 0; j < 2; ++j) {
            const int n = bn0 + wn * 32 + j * 16 + lrow;
            if (n < N) {
                const float bv = bias ? bias[n] : 0.0f;
                #pragma unroll
                for (int r = 0; r < 8; ++r) {
                    const int m = mbase + r;
                    if (m < M) {
                        float v = acc[i][j][r] * alpha + bv;
                        if (resid) v += resid[(size_t)m * ldc + n];
                        if (swish) v = v / (1.0f + __expf(-v));
                        if (outF)  outF[(size_t)m * ldc + n] = v;
                        if (outB)  outB[(size_t)m * ldcb + n] = f2bf(v);
                    }
                }
            }
        }
    }
}

// ---------------------------------------------------------------------------
// Tiled transposes (32x32 tiles via LDS, coalesced loads and stores)
// out[c*ldo + r] = in[r*ldi + c]; rows c in [Cvalid, Cout) are zero-filled.
// ---------------------------------------------------------------------------
__global__ __launch_bounds__(256)
void transpose_f2b(const float* __restrict__ in, __bf16* __restrict__ out,
                   int R, int Cvalid, int Cout, int ldi, int ldo)
{
    __shared__ float tile[32][33];
    const int bx = blockIdx.x * 32;   // c base
    const int by = blockIdx.y * 32;   // r base
    for (int i = threadIdx.y; i < 32; i += 8) {
        int r = by + i, c = bx + threadIdx.x;
        tile[i][threadIdx.x] = (r < R && c < Cvalid) ? in[(size_t)r * ldi + c] : 0.0f;
    }
    __syncthreads();
    for (int i = threadIdx.y; i < 32; i += 8) {
        int c = bx + i, r = by + threadIdx.x;
        if (c < Cout && r < R) out[(size_t)c * ldo + r] = f2bf(tile[threadIdx.x][i]);
    }
}

__global__ __launch_bounds__(256)
void transpose_b2b(const __bf16* __restrict__ in, __bf16* __restrict__ out,
                   int R, int C, int ldi, int ldo)
{
    __shared__ __bf16 tile[32][34];
    const int bx = blockIdx.x * 32;   // c base
    const int by = blockIdx.y * 32;   // r base
    for (int i = threadIdx.y; i < 32; i += 8) {
        int r = by + i, c = bx + threadIdx.x;
        if (r < R && c < C) tile[i][threadIdx.x] = in[(size_t)r * ldi + c];
    }
    __syncthreads();
    for (int i = threadIdx.y; i < 32; i += 8) {
        int c = bx + i, r = by + threadIdx.x;
        if (c < C && r < R) out[(size_t)c * ldo + r] = tile[threadIdx.x][i];
    }
}

// ---------------------------------------------------------------------------
// Causal row softmax: P[row, c] = softmax(S[row, 0..row])[c], 0 for c > row
// ---------------------------------------------------------------------------
__global__ __launch_bounds__(256)
void softmax_causal(const float* __restrict__ S, __bf16* __restrict__ P, int n)
{
    __shared__ float red[256];
    const int row = blockIdx.x;
    const int tid = threadIdx.x;
    const int L = row + 1;
    const float* srow = S + (size_t)row * n;

    float mx = -3.4e38f;
    for (int c = tid; c < L; c += 256) mx = fmaxf(mx, srow[c]);
    red[tid] = mx; __syncthreads();
    for (int s = 128; s > 0; s >>= 1) { if (tid < s) red[tid] = fmaxf(red[tid], red[tid + s]); __syncthreads(); }
    mx = red[0]; __syncthreads();

    float sum = 0.0f;
    for (int c = tid; c < L; c += 256) sum += __expf(srow[c] - mx);
    red[tid] = sum; __syncthreads();
    for (int s = 128; s > 0; s >>= 1) { if (tid < s) red[tid] += red[tid + s]; __syncthreads(); }
    const float inv = 1.0f / red[0];

    __bf16* prow = P + (size_t)row * n;
    for (int c = tid; c < n; c += 256) {
        float v = (c < L) ? __expf(srow[c] - mx) * inv : 0.0f;
        prow[c] = f2bf(v);
    }
}

// ---------------------------------------------------------------------------
// LayerNorm (no scale/bias), one block per row
// ---------------------------------------------------------------------------
__global__ __launch_bounds__(256)
void layernorm_rows(const float* __restrict__ X, __bf16* __restrict__ Y, int d)
{
    __shared__ float r1[256], r2[256];
    const int row = blockIdx.x;
    const int tid = threadIdx.x;
    const float* x = X + (size_t)row * d;
    float s1 = 0.0f, s2 = 0.0f;
    for (int c = tid; c < d; c += 256) { float v = x[c]; s1 += v; s2 += v * v; }
    r1[tid] = s1; r2[tid] = s2; __syncthreads();
    for (int s = 128; s > 0; s >>= 1) {
        if (tid < s) { r1[tid] += r1[tid + s]; r2[tid] += r2[tid + s]; }
        __syncthreads();
    }
    const float mean = r1[0] / d;
    const float var  = r2[0] / d - mean * mean;
    const float rstd = rsqrtf(var + EPSLN);
    __bf16* y = Y + (size_t)row * d;
    for (int c = tid; c < d; c += 256) y[c] = f2bf((x[c] - mean) * rstd);
}

// ---------------------------------------------------------------------------
// Embedding gather + interleaved RoPE over full d_model -> bf16 h
// ---------------------------------------------------------------------------
__global__ __launch_bounds__(256)
void embed_rope(const int* __restrict__ x, const float* __restrict__ emb,
                __bf16* __restrict__ hB)
{
    const int idx = blockIdx.x * 256 + threadIdx.x;   // SEQ*(DM/2)
    const int s = idx / (DM / 2);
    const int j = idx % (DM / 2);
    const int tok = x[s];
    const float x0 = emb[(size_t)tok * DM + 2 * j];
    const float x1 = emb[(size_t)tok * DM + 2 * j + 1];
    const float inv = __expf(-(2.0f * j / (float)DM) * 9.210340371976184f); // ln(10000)
    float sn, cs;
    __sincosf((float)s * inv, &sn, &cs);
    hB[(size_t)s * DM + 2 * j]     = f2bf(x0 * cs - x1 * sn);
    hB[(size_t)s * DM + 2 * j + 1] = f2bf(x1 * cs + x0 * sn);
}

// ---------------------------------------------------------------------------
// w_out[l].reshape(H,d,d).sum(axis=0) -> f32 (coalesced); transposed later
// ---------------------------------------------------------------------------
__global__ __launch_bounds__(256)
void sum_heads(const float* __restrict__ w_out, float* __restrict__ wsumF)
{
    const int idx = blockIdx.x * 256 + threadIdx.x;   // NL*DM*DM
    const int l = idx / (DM * DM);
    const int r = idx % (DM * DM);
    const int i = r / DM, j = r % DM;
    float s = 0.0f;
    #pragma unroll
    for (int h = 0; h < NH; ++h)
        s += w_out[((size_t)l * NH * DM + (size_t)h * DM + i) * DM + j];
    wsumF[(size_t)l * DM * DM + r] = s;
}

// ---------------------------------------------------------------------------
// Launcher. Workspace need: ~112 MB.
// ---------------------------------------------------------------------------
extern "C" void kernel_launch(void* const* d_in, const int* in_sizes, int n_in,
                              void* d_out, int out_size, void* d_ws, size_t ws_size,
                              hipStream_t stream)
{
    (void)in_sizes; (void)n_in; (void)out_size; (void)ws_size;
    const int*   x     = (const int*)d_in[0];
    const float* emb   = (const float*)d_in[1];
    const float* wi    = (const float*)d_in[2];
    const float* w_out = (const float*)d_in[3];
    const float* b_out = (const float*)d_in[4];
    const float* w_lm  = (const float*)d_in[5];
    const float* b_lm  = (const float*)d_in[6];
    float* out = (float*)d_out;

    char* p = (char*)d_ws;
    auto carve = [&](size_t bytes) { char* r = p; p += (bytes + 255) & ~(size_t)255; return r; };
    __bf16* hB    = (__bf16*)carve((size_t)SEQ * DM * 2);    // layer input (bf16)
    float*  qF    = (float*) carve((size_t)SEQ * DM * 4);    // q (f32, residual)
    __bf16* qB    = (__bf16*)carve((size_t)SEQ * DM * 2);    // q (bf16)
    __bf16* qTB   = (__bf16*)carve((size_t)DM * SEQ * 2);    // q^T (bf16)
    float*  Sf    = (float*) carve((size_t)SEQ * SEQ * 4);   // attention scores
    __bf16* Pb    = (__bf16*)carve((size_t)SEQ * SEQ * 2);   // softmax probs
    __bf16* headB = (__bf16*)carve((size_t)SEQ * DM * 2);    // q + P@q
    float*  tF    = (float*) carve((size_t)SEQ * DM * 4);    // out-proj (pre-LN)
    __bf16* lnB   = (__bf16*)carve((size_t)SEQ * DM * 2);    // LN output
    __bf16* swB   = (__bf16*)carve((size_t)SEQ * DM * 2);    // swish(h@w)
    float*  wsumF = (float*) carve((size_t)NL * DM * DM * 4);
    __bf16* wiT   = (__bf16*)carve((size_t)NL * DM * DM * 2);  // wi^T  [k-major]
    __bf16* wsT   = (__bf16*)carve((size_t)NL * DM * DM * 2);  // wsum^T
    __bf16* wlmT  = (__bf16*)carve((size_t)VOCP * DM * 2);     // w_lm^T [VOCP,DM]

    // ---- precompute ----
    dim3 tb(32, 8);
    embed_rope<<<SEQ * (DM / 2) / 256, 256, 0, stream>>>(x, emb, hB);
    sum_heads <<<(NL * DM * DM) / 256, 256, 0, stream>>>(w_out, wsumF);
    for (int l = 0; l < NL; ++l) {
        transpose_f2b<<<dim3(DM / 32, DM / 32), tb, 0, stream>>>(
            wi    + (size_t)l * DM * DM, wiT + (size_t)l * DM * DM, DM, DM, DM, DM, DM);
        transpose_f2b<<<dim3(DM / 32, DM / 32), tb, 0, stream>>>(
            wsumF + (size_t)l * DM * DM, wsT + (size_t)l * DM * DM, DM, DM, DM, DM, DM);
    }
    transpose_f2b<<<dim3(VOCP / 32, DM / 32), tb, 0, stream>>>(
        w_lm, wlmT, DM, VOC, VOCP, VOC, DM);

    const float scale = 0.044194173824159216f;  // 1/sqrt(512)
    dim3 blk(256);
    dim3 gDD((DM  + 127) / 128, (SEQ + 127) / 128);
    dim3 gSS((SEQ + 127) / 128, (SEQ + 127) / 128);
    dim3 gLM((VOC + 127) / 128, (SEQ + 127) / 128);

    for (int l = 0; l < NL; ++l) {
        const __bf16* wl  = wiT + (size_t)l * DM * DM;
        const __bf16* wol = wsT + (size_t)l * DM * DM;
        const float*  bl  = b_out + (size_t)l * DM;

        // q = h @ wi[l]  (f32 + bf16)
        gemm_bf16<<<gDD, blk, 0, stream>>>(hB, wl, qF, qB, nullptr, nullptr,
            SEQ, DM, DM, DM, DM, DM, DM, 1.0f, 0, 0);
        // q^T for the P@q GEMM
        transpose_b2b<<<dim3(DM / 32, SEQ / 32), tb, 0, stream>>>(qB, qTB, SEQ, DM, DM, SEQ);
        // S = scale * q @ q^T  (causal block skip)
        gemm_bf16<<<gSS, blk, 0, stream>>>(qB, qB, Sf, nullptr, nullptr, nullptr,
            SEQ, SEQ, DM, DM, DM, SEQ, SEQ, scale, 0, 1);
        // P = causal softmax(S)
        softmax_causal<<<SEQ, blk, 0, stream>>>(Sf, Pb, SEQ);
        // head = P @ q + q
        gemm_bf16<<<gDD, blk, 0, stream>>>(Pb, qTB, nullptr, headB, nullptr, qF,
            SEQ, DM, SEQ, SEQ, SEQ, DM, DM, 1.0f, 0, 0);
        // t = head @ sum_heads(w_out[l]) + b_out[l]
        gemm_bf16<<<gDD, blk, 0, stream>>>(headB, wol, tF, nullptr, bl, nullptr,
            SEQ, DM, DM, DM, DM, DM, DM, 1.0f, 0, 0);
        // LayerNorm
        layernorm_rows<<<SEQ, blk, 0, stream>>>(tF, lnB, DM);
        // sw = swish(ln @ wi[l])
        gemm_bf16<<<gDD, blk, 0, stream>>>(lnB, wl, nullptr, swB, nullptr, nullptr,
            SEQ, DM, DM, DM, DM, DM, DM, 1.0f, 1, 0);
        // h = sw @ wi[l]
        gemm_bf16<<<gDD, blk, 0, stream>>>(swB, wl, nullptr, hB, nullptr, nullptr,
            SEQ, DM, DM, DM, DM, DM, DM, 1.0f, 0, 0);
    }

    // logits = h @ w_lm + b_lm  (f32 to d_out)
    gemm_bf16<<<gLM, blk, 0, stream>>>(hB, wlmT, out, nullptr, b_lm, nullptr,
        SEQ, VOC, DM, DM, DM, VOC, VOC, 1.0f, 0, 0);
}
